// MultiHeadSelfAttention_14199161880718
// MI455X (gfx1250) — matrix operand order repaired and checked by reading, not compiled, not run on hardware
//
#include <hip/hip_runtime.h>
#include <hip/hip_bf16.h>

// ---------------------------------------------------------------------------
// MI455X (gfx1250, wave32) multi-head self-attention forward.
//   x[8,1024,1024] f32 -> out[8,1024,1024] f32
// All GEMM-shaped math runs on v_wmma_f32_16x16x32_bf16 (f32 accumulate).
// Layouts are chosen so every WMMA fragment load is a b128 access:
//   - weights pre-transposed to [N,K] bf16
//   - V stored d-major [B*H][64][1024]
// GEMM tiles stream through LDS with global_load_async_to_lds_b128 in a
// ping-pong double buffer: s_wait_asynccnt 5 retires the current buffer's
// (in-order) loads while the next tile's DMA overlaps the WMMAs.
// ---------------------------------------------------------------------------

typedef __bf16 bf16_t;
typedef __attribute__((ext_vector_type(16))) __bf16 bf16x16;
typedef __attribute__((ext_vector_type(8)))  float  v8f;

#define WMMA_BF16(a, b, c)                                                    \
  __builtin_amdgcn_wmma_f32_16x16x32_bf16(false, (a), false, (b), (short)0,   \
                                          (c), false, false)

union frag_u { uint4 u4[2]; bf16x16 v; };
union pack8_u { uint4 u; __bf16 e[8]; };

// ---- CDNA5 async global->LDS copy (ASYNCcnt), with safe fallback ----------
#if defined(__AMDGCN__) && __has_builtin(__builtin_amdgcn_global_load_async_to_lds_b128)
#define USE_ASYNC_LDS 1
#else
#define USE_ASYNC_LDS 0
#endif

#if USE_ASYNC_LDS
// Builtin signature (probe-verified): (v4i AS1*, v4i AS3*, imm, imm).
// generic->AS3: low 32 bits of a generic LDS pointer are the LDS offset.
typedef __attribute__((__vector_size__(16))) int v4i_t;
#define ASYNC_CP_B128(gsrc, ldst)                                             \
  __builtin_amdgcn_global_load_async_to_lds_b128(                             \
      (__attribute__((address_space(1))) v4i_t*)(unsigned long long)(const void*)(gsrc), \
      (__attribute__((address_space(3))) v4i_t*)(unsigned)(unsigned long long)(const void*)(ldst), \
      0, 0)
#if __has_builtin(__builtin_amdgcn_s_wait_asynccnt)
#define ASYNC_WAIT(n) __builtin_amdgcn_s_wait_asynccnt(n)
#else
#define ASYNC_WAIT(n) asm volatile("s_wait_asynccnt " #n ::: "memory")
#endif
#else
#define ASYNC_CP_B128(gsrc, ldst) (*(uint4*)(ldst) = *(const uint4*)(gsrc))
#define ASYNC_WAIT(n) ((void)0)
#endif

// ---------------------------------------------------------------------------
// A-matrix fragment, 16x32 bf16 (ISA 7.12.2): lane L holds row (L&15);
// elements 0..7 -> K = (L>=16 ? 8 : 0)+e, elements 8..15 -> same +16.
// Two contiguous 16-byte chunks per lane -> two b128 loads.
__device__ __forceinline__ bf16x16 load_A_frag(const bf16_t* base, int ld,
                                               int lane) {
  const bf16_t* p = base + (size_t)(lane & 15) * ld + ((lane & 16) >> 1);
  frag_u f;
  f.u4[0] = *(const uint4*)(p);
  f.u4[1] = *(const uint4*)(p + 16);
  return f.v;
}

// B-matrix fragment from a row-major [N x K] source (logical B = src^T):
// lane L holds column n=(L&15); elements 0..7 -> K = k0 + (L>=16 ? 8 : 0)+e, elements 8..15 -> same +16.
// two contiguous 8-bf16 chunks per lane -> two b128 loads.
__device__ __forceinline__ bf16x16 load_Bt_frag(const bf16_t* base, int ld,
                                                int k0, int lane) {
  const bf16_t* p = base + (size_t)(lane & 15) * ld + k0 + ((lane & 16) >> 1);
  frag_u f;
  f.u4[0] = *(const uint4*)(p);
  f.u4[1] = *(const uint4*)(p + 16);
  return f.v;
}

// ---------------------------------------------------------------------------
__global__ void cast_f32_to_bf16(const float* __restrict__ in,
                                 bf16_t* __restrict__ out, int n) {
  int i = blockIdx.x * blockDim.x + threadIdx.x;
  int stride = gridDim.x * blockDim.x;
  for (; i < n; i += stride) out[i] = (bf16_t)in[i];
}

// in [R][C] f32 -> out [C][R] bf16, 32x32 LDS tiles (coalesced both sides).
__global__ __launch_bounds__(256) void transpose_cast_f32_bf16(
    const float* __restrict__ in, bf16_t* __restrict__ out, int R, int C) {
  __shared__ float tile[32][33];
  const int c0 = blockIdx.x * 32, r0 = blockIdx.y * 32;
  const int tx = threadIdx.x & 31, ty = threadIdx.x >> 5;  // 32 x 8
#pragma unroll
  for (int i = 0; i < 32; i += 8)
    tile[ty + i][tx] = in[(size_t)(r0 + ty + i) * C + c0 + tx];
  __syncthreads();
#pragma unroll
  for (int i = 0; i < 32; i += 8)
    out[(size_t)(c0 + ty + i) * R + r0 + tx] = (bf16_t)tile[tx][ty + i];
}

// ---------------------------------------------------------------------------
// Tiled WMMA GEMM: C[M,N] = A[M,K] * Bt^T + bias   (Bt is [N,K] row-major).
// Block = 256 threads = 8 waves (2 M x 4 N), block tile 64x256, K step 32,
// per-wave tile 32x64 (2x4 WMMA grid, f32 accumulate).
// Double-buffered async-to-LDS pipeline: 5 async b128 ops per thread per
// tile; ASYNC_WAIT(5) retires the current buffer while the next streams in.
// mode 0: scatter bf16 into Q/K [B*H][1024][64] and V^T [B*H][64][1024],
//         Q scaled by qscale.   mode 1: f32 store with bias (projection).
__global__ __launch_bounds__(256) void wmma_gemm_bf16(
    const bf16_t* __restrict__ A, const bf16_t* __restrict__ Bt,
    const float* __restrict__ bias, bf16_t* __restrict__ out_qkv,
    float* __restrict__ out_f32, int M, int N, int K, int mode, float qscale) {
  __shared__ bf16_t As[2][64][32];     // 2 x 4 KB
  __shared__ bf16_t Bts[2][256][32];   // 2 x 16 KB
  const int t    = threadIdx.x;
  const int lane = t & 31;
  const int wave = t >> 5;
  const int wm   = wave >> 2;       // 0..1
  const int wn   = wave & 3;        // 0..3
  const int m0   = blockIdx.y * 64;
  const int n0   = blockIdx.x * 256;

  v8f acc[2][4] = {};

  const int a_row = t >> 2, a_col = (t & 3) * 8;  // A: 64x32, 8 bf16/thread
  // B: 256 rows x 32 k, one row (32 bf16 = 4 b128) per thread.

  auto issue_tile = [&](int buf, int kt) {
    ASYNC_CP_B128(A + (size_t)(m0 + a_row) * K + kt + a_col,
                  &As[buf][a_row][a_col]);
    const bf16_t* gb = Bt + (size_t)(n0 + t) * K + kt;
#pragma unroll
    for (int c = 0; c < 4; ++c)
      ASYNC_CP_B128(gb + c * 8, &Bts[buf][t][c * 8]);
  };

  issue_tile(0, 0);
  int cur = 0;
  for (int kt = 0; kt < K; kt += 32) {
    if (kt + 32 < K) {
      issue_tile(cur ^ 1, kt + 32);  // next tile DMA overlaps this compute
      ASYNC_WAIT(5);                 // in-order: current buffer's 5 retired
    } else {
      ASYNC_WAIT(0);
    }
    __syncthreads();

    bf16x16 a0 = load_A_frag(&As[cur][wm * 32][0], 32, lane);
    bf16x16 a1 = load_A_frag(&As[cur][wm * 32 + 16][0], 32, lane);
#pragma unroll
    for (int j = 0; j < 4; ++j) {
      bf16x16 bj = load_Bt_frag(&Bts[cur][wn * 64 + j * 16][0], 32, 0, lane);
      acc[0][j] = WMMA_BF16(a0, bj, acc[0][j]);
      acc[1][j] = WMMA_BF16(a1, bj, acc[1][j]);
    }
    __syncthreads();  // compute done before this buffer is refilled
    cur ^= 1;
  }

  // C/D layout: VGPR v -> row = v + 8*(lane>=16); col = lane&15.
  const int hi8 = (lane & 16) ? 8 : 0;
  const int cn  = lane & 15;
#pragma unroll
  for (int i = 0; i < 2; ++i)
#pragma unroll
    for (int j = 0; j < 4; ++j) {
      const int col = n0 + wn * 64 + j * 16 + cn;
      const float bv = bias[col];
      if (mode == 0) {
        const int sel  = col >> 10;          // 0=Q 1=K 2=V
        const int head = (col >> 6) & 15;
        const int d    = col & 63;
        const float sc = (sel == 0) ? qscale : 1.0f;
        bf16_t* seg = out_qkv + (size_t)sel * (size_t)(128 * 1024 * 64);
#pragma unroll
        for (int v = 0; v < 8; ++v) {
          const int r = m0 + wm * 32 + i * 16 + v + hi8;  // global token row
          const int batch = r >> 10, tok = r & 1023;
          const bf16_t val = (bf16_t)((acc[i][j][v] + bv) * sc);
          if (sel == 2)  // V stored d-major for contiguous P*V B-fragments
            seg[(((size_t)(batch * 16 + head)) * 64 + d) * 1024 + tok] = val;
          else
            seg[(((size_t)(batch * 16 + head)) * 1024 + tok) * 64 + d] = val;
        }
      } else {
#pragma unroll
        for (int v = 0; v < 8; ++v) {
          const int r = m0 + wm * 32 + i * 16 + v + hi8;
          out_f32[(size_t)r * N + col] = acc[i][j][v] + bv;
        }
      }
    }
}

// ---------------------------------------------------------------------------
// Attention: one block handles 128 query rows of one (batch, head).
// 8 waves; wave w owns 16 q-rows. Full 16x1024 score row lives in LDS
// (256 KB of CDNA5's 320 KB WGP LDS) -> simple two-pass softmax.
// Scale pre-folded into Q. Output fuses [B,H,N,hd] -> [B,N,D] transpose.
__global__ __launch_bounds__(256) void attention_wmma(
    const bf16_t* __restrict__ Q, const bf16_t* __restrict__ Km,
    const bf16_t* __restrict__ Vt, bf16_t* __restrict__ O) {
  extern __shared__ char smem_raw[];
  bf16_t* S = (bf16_t*)smem_raw;  // [8 waves][16 rows][1024 keys]
  __shared__ float rowsum[128];

  const int bh   = blockIdx.x >> 3;  // batch*16 + head
  const int qblk = blockIdx.x & 7;
  const int t    = threadIdx.x;
  const int lane = t & 31;
  const int wave = t >> 5;
  const int batch = bh >> 4, head = bh & 15;

  const bf16_t* Qb  = Q  + (size_t)bh * 1024 * 64;
  const bf16_t* Kb  = Km + (size_t)bh * 1024 * 64;
  const bf16_t* Vtb = Vt + (size_t)bh * 64 * 1024;  // [64 d][1024 keys]
  const int q0 = qblk * 128 + wave * 16;
  bf16_t* Sw = S + (size_t)wave * 16 * 1024;

  const int hi8 = (lane & 16) ? 8 : 0;
  const int cn  = lane & 15;

  // Q fragments (16 rows x 64 dim) held in registers for the whole kernel.
  bf16x16 a0 = load_A_frag(Qb + (size_t)q0 * 64, 64, lane);
  bf16x16 a1 = load_A_frag(Qb + (size_t)q0 * 64 + 32, 64, lane);

  // ---- S = Q * K^T : 64 key-tiles of 16, K-dim 64 = 2 WMMAs each ----
  for (int kb = 0; kb < 64; ++kb) {
    const bf16_t* kt = Kb + (size_t)(kb * 16) * 64;
    v8f c = {};
    c = WMMA_BF16(a0, load_Bt_frag(kt, 64, 0, lane), c);
    c = WMMA_BF16(a1, load_Bt_frag(kt, 64, 32, lane), c);
    bf16_t* sp = Sw + kb * 16 + cn;
#pragma unroll
    for (int v = 0; v < 8; ++v) sp[(size_t)(v + hi8) * 1024] = (bf16_t)c[v];
  }
  asm volatile("s_wait_dscnt 0" ::: "memory");  // LDS RAW across lanes

  // ---- softmax: 2 lanes per row (512 keys each), b128-vectorized ----
  {
    const int row  = cn;
    const int half = (lane >> 4) & 1;
    bf16_t* sr = Sw + (size_t)row * 1024 + half * 512;
    float m = -3.0e38f;
    for (int i = 0; i < 512; i += 8) {
      pack8_u p; p.u = *(const uint4*)(sr + i);
#pragma unroll
      for (int e = 0; e < 8; ++e) m = fmaxf(m, (float)p.e[e]);
    }
    m = fmaxf(m, __shfl_xor(m, 16, 32));
    float sum = 0.0f;
    for (int i = 0; i < 512; i += 8) {
      pack8_u p; p.u = *(const uint4*)(sr + i);
#pragma unroll
      for (int e = 0; e < 8; ++e) {
        float pe = __expf((float)p.e[e] - m);
        sum += pe;
        p.e[e] = (bf16_t)pe;
      }
      *(uint4*)(sr + i) = p.u;  // P overwrites S in place (disjoint per lane)
    }
    sum += __shfl_xor(sum, 16, 32);
    if (half == 0) rowsum[wave * 16 + row] = sum;
  }
  asm volatile("s_wait_dscnt 0" ::: "memory");

  // ---- O = P * V : K-dim 1024 in steps of 32, N-dim 64 = 4 tiles ----
  // V^T layout makes every B-fragment two contiguous b128 loads.
  v8f co[4] = {};
  for (int kb = 0; kb < 32; ++kb) {
    bf16x16 ap = load_A_frag(Sw + kb * 32, 1024, lane);
#pragma unroll
    for (int nt = 0; nt < 4; ++nt)
      co[nt] = WMMA_BF16(
          ap, load_Bt_frag(Vtb + (size_t)(nt * 16) * 1024, 1024, kb * 32, lane),
          co[nt]);
  }

  // ---- normalize by row sum, scatter to [token, head*64+d] ----
#pragma unroll
  for (int nt = 0; nt < 4; ++nt) {
    const int d = nt * 16 + cn;
#pragma unroll
    for (int v = 0; v < 8; ++v) {
      const int row = v + hi8;
      const float inv = 1.0f / rowsum[wave * 16 + row];
      const size_t r = (size_t)batch * 1024 + (q0 + row);
      O[r * 1024 + head * 64 + d] = (bf16_t)(co[nt][v] * inv);
    }
  }
}

// ---------------------------------------------------------------------------
extern "C" void kernel_launch(void* const* d_in, const int* in_sizes, int n_in,
                              void* d_out, int out_size, void* d_ws,
                              size_t ws_size, hipStream_t stream) {
  const float* x      = (const float*)d_in[0];
  const float* w_qkv  = (const float*)d_in[1];
  const float* b_qkv  = (const float*)d_in[2];
  const float* w_proj = (const float*)d_in[3];
  const float* b_proj = (const float*)d_in[4];
  float* out = (float*)d_out;

  const int M = 8 * 1024;  // tokens
  const int D = 1024;
  const float scale = 0.125f;  // 64^-0.5

  // Workspace layout (bf16 staging, ~92 MB total)
  char* ws = (char*)d_ws;
  bf16_t* Xbf    = (bf16_t*)ws;  ws += (size_t)M * D * 2;
  bf16_t* WqkvT  = (bf16_t*)ws;  ws += (size_t)3 * D * D * 2;  // [3072][1024]
  bf16_t* WprojT = (bf16_t*)ws;  ws += (size_t)D * D * 2;      // [1024][1024]
  bf16_t* QKV    = (bf16_t*)ws;  ws += (size_t)3 * 128 * 1024 * 64 * 2;
  bf16_t* Attn   = (bf16_t*)ws;  // M * D bf16
  const size_t seg = (size_t)128 * 1024 * 64;

  cast_f32_to_bf16<<<4096, 256, 0, stream>>>(x, Xbf, M * D);
  {  // w_qkv [1024][3072] -> WqkvT [3072][1024] bf16
    dim3 g(3 * D / 32, D / 32);
    transpose_cast_f32_bf16<<<g, 256, 0, stream>>>(w_qkv, WqkvT, D, 3 * D);
  }
  {  // w_proj [1024][1024] -> WprojT [1024][1024] bf16
    dim3 g(D / 32, D / 32);
    transpose_cast_f32_bf16<<<g, 256, 0, stream>>>(w_proj, WprojT, D, D);
  }

  // QKV projection: [8192,1024] x [1024,3072] (+bias, +scale, head split)
  {
    dim3 grid(3 * D / 256, M / 64);
    wmma_gemm_bf16<<<grid, 256, 0, stream>>>(Xbf, WqkvT, b_qkv, QKV, nullptr,
                                             M, 3 * D, D, 0, scale);
  }

  // Attention: 128 (b,h) pairs x 8 q-blocks; 256 KB dynamic LDS.
  {
    static_cast<void>(hipFuncSetAttribute(
        (const void*)attention_wmma,
        hipFuncAttributeMaxDynamicSharedMemorySize, 256 * 1024));
    attention_wmma<<<1024, 256, 256 * 1024, stream>>>(QKV, QKV + seg,
                                                      QKV + 2 * seg, Attn);
  }

  // Output projection: [8192,1024] x [1024,1024] -> f32 out
  {
    dim3 grid(D / 256, M / 64);
    wmma_gemm_bf16<<<grid, 256, 0, stream>>>(Attn, WprojT, b_proj, nullptr, out,
                                             M, D, D, 1, 0.0f);
  }
}
